// SelectiveSSMBlock_43937515438700
// MI455X (gfx1250) — matrix operand order repaired and checked by reading, not compiled, not run on hardware
//
#include <hip/hip_runtime.h>
#include <hip/hip_bf16.h>

// ---------------------------------------------------------------------------
// SelectiveSSMBlock for MI455X (gfx1250, wave32, WMMA)
//   B=2, T=2048, D_MODEL=1024, D_INNER=2048, D_STATE=32, DT_RANK=64
// ---------------------------------------------------------------------------

#define BB      2
#define TT      2048
#define DM      1024
#define EE      2048          // D_INNER
#define NS      32            // D_STATE (== wave32!)
#define DTR     64            // DT_RANK
#define MROWS   (BB * TT)     // 4096 token rows

typedef __attribute__((ext_vector_type(16))) _Float16 v16h;
typedef __attribute__((ext_vector_type(8)))  _Float16 v8h;
typedef __attribute__((ext_vector_type(8)))  float    v8f;

union hv16 { v16h v; v8h h[2]; };

// ---------------------------------------------------------------------------
// f32 -> f16 elementwise conversion (weights)
// ---------------------------------------------------------------------------
__global__ void cvt_f32_f16_kernel(const float* __restrict__ in,
                                   _Float16* __restrict__ out, size_t n) {
    size_t i = (size_t)blockIdx.x * blockDim.x + threadIdx.x;
    if (i < n) out[i] = (_Float16)in[i];
}

// ---------------------------------------------------------------------------
// LayerNorm over D_MODEL=1024, one 256-thread block (8 waves) per token row.
// Writes f16 normalized activations for the in-proj WMMA GEMM.
// ---------------------------------------------------------------------------
__global__ __launch_bounds__(256)
void layernorm_kernel(const float* __restrict__ x,
                      const float* __restrict__ g,
                      const float* __restrict__ b,
                      _Float16* __restrict__ xn) {
    int row = blockIdx.x;
    const float* xr = x + (size_t)row * DM;
    __shared__ float red0[8], red1[8];
    float s = 0.f, s2 = 0.f;
    for (int i = threadIdx.x; i < DM; i += 256) {
        float v = xr[i]; s += v; s2 += v * v;
    }
    for (int off = 16; off >= 1; off >>= 1) {
        s  += __shfl_xor(s,  off, 32);
        s2 += __shfl_xor(s2, off, 32);
    }
    int wave = threadIdx.x >> 5, lane = threadIdx.x & 31;
    if (lane == 0) { red0[wave] = s; red1[wave] = s2; }
    __syncthreads();
    if (threadIdx.x == 0) {
        float a = 0.f, c = 0.f;
        for (int w = 0; w < 8; ++w) { a += red0[w]; c += red1[w]; }
        red0[0] = a; red1[0] = c;
    }
    __syncthreads();
    float mu  = red0[0] * (1.0f / DM);
    float var = red1[0] * (1.0f / DM) - mu * mu;
    float rs  = rsqrtf(var + 1e-5f);
    for (int i = threadIdx.x; i < DM; i += 256) {
        float v = (xr[i] - mu) * rs * g[i] + b[i];
        xn[(size_t)row * DM + i] = (_Float16)v;
    }
}

// ---------------------------------------------------------------------------
// Fragment loads per CDNA5 ISA 7.12.2 (16-bit A 16x32, B 32x16; wave32).
//   A: lane(M=r, h) holds K-runs [8h,8h+8) and [16+8h,16+8h+8)  -> 2x b128
//   B: lane(N=r, h) holds K-runs [8h,8h+8) and [16+8h,16+8h+8) of column N,
//      from weight row (n0+j*16+r)                              -> 2x b128
// ---------------------------------------------------------------------------
template<int NSUB>
__device__ __forceinline__ void load_frags(const _Float16* __restrict__ arow,
                                           const _Float16* __restrict__ bbase,
                                           int K, int k0, int half,
                                           hv16& a, hv16 (&b)[NSUB]) {
    a.h[0] = *(const v8h*)(arow + k0 + half * 8);
    a.h[1] = *(const v8h*)(arow + k0 + 16 + half * 8);
#pragma unroll
    for (int j = 0; j < NSUB; ++j) {
        const _Float16* brow = bbase + (size_t)j * 16 * K + k0 + half * 8;
        b[j].h[0] = *(const v8h*)(brow);
        b[j].h[1] = *(const v8h*)(brow + 16);
    }
}

// ---------------------------------------------------------------------------
// WMMA f16 GEMM: C[M,N] = A[M,K] * Bw[N,K]^T  (Bw = row-major weight (out,in))
// One wave computes a 16 x (16*NSUB) output strip; K-loop in steps of 32,
// software-pipelined x2 with ping-pong register fragments so the next
// k-step's global_load_b128 clause overlaps the current WMMAs.
// EPI: 0 = plain f32 store
//      1 = +bias then softplus                       (dt projection)
//      2 = f32 store + f16 copy of cols<64           (x_dbl -> dt_low)
//      3 = + residual (aux, ld=N)                    (out projection)
//      4 = f32 store + f16 copy of cols<EE           (in-proj -> x_ssm f16)
// ---------------------------------------------------------------------------
template<int NSUB, int EPI>
__global__ __launch_bounds__(256)
void gemm_wmma_kernel(const _Float16* __restrict__ A,
                      const _Float16* __restrict__ Bw,
                      float* __restrict__ C,
                      int M, int N, int K,
                      const float* __restrict__ aux,
                      _Float16* __restrict__ aux16) {
    int wave = threadIdx.x >> 5;
    int lane = threadIdx.x & 31;
    int gw   = blockIdx.x * (blockDim.x >> 5) + wave;
    int NG   = N / (16 * NSUB);
    int mt   = gw / NG;
    int ng   = gw - mt * NG;
    if (mt * 16 >= M) return;               // wave-uniform (EXEC stays all-1s)
    int m0 = mt * 16;
    int n0 = ng * 16 * NSUB;
    int half = lane >> 4;                   // 0: lanes 0-15, 1: lanes 16-31
    int r    = lane & 15;

    v8f acc[NSUB];
#pragma unroll
    for (int j = 0; j < NSUB; ++j) acc[j] = v8f{0.f,0.f,0.f,0.f,0.f,0.f,0.f,0.f};

    const _Float16* arow  = A  + (size_t)(m0 + r) * K;
    const _Float16* bbase = Bw + (size_t)(n0 + r) * K;

    // ping-pong fragment buffers; K is always a multiple of 64 here
    hv16 aP, bP[NSUB], aQ, bQ[NSUB];
    load_frags<NSUB>(arow, bbase, K, 0, half, aP, bP);

    for (int k0 = 0; k0 < K; k0 += 64) {
        load_frags<NSUB>(arow, bbase, K, k0 + 32, half, aQ, bQ);
#pragma unroll
        for (int j = 0; j < NSUB; ++j)
            acc[j] = __builtin_amdgcn_wmma_f32_16x16x32_f16(
                false, aP.v, false, bP[j].v, (short)0, acc[j], false, false);
        if (k0 + 64 < K)
            load_frags<NSUB>(arow, bbase, K, k0 + 64, half, aP, bP);
#pragma unroll
        for (int j = 0; j < NSUB; ++j)
            acc[j] = __builtin_amdgcn_wmma_f32_16x16x32_f16(
                false, aQ.v, false, bQ[j].v, (short)0, acc[j], false, false);
    }

    // C/D layout: VGPR p, lane -> row = m0 + p + 8*half, col = n0 + j*16 + r
#pragma unroll
    for (int j = 0; j < NSUB; ++j) {
        int col = n0 + j * 16 + r;
#pragma unroll
        for (int p = 0; p < 8; ++p) {
            int row = m0 + p + 8 * half;
            float v = acc[j][p];
            if (EPI == 1) {
                v += aux[col];
                v = (v > 20.0f) ? v : __logf(1.0f + __expf(v));   // softplus
                C[(size_t)row * N + col] = v;
            } else if (EPI == 2) {
                C[(size_t)row * N + col] = v;
                if (col < DTR) aux16[(size_t)row * DTR + col] = (_Float16)v;
            } else if (EPI == 3) {
                C[(size_t)row * N + col] = v + aux[(size_t)row * N + col];
            } else if (EPI == 4) {
                C[(size_t)row * N + col] = v;
                if (col < EE) aux16[(size_t)row * EE + col] = (_Float16)v;
            } else {
                C[(size_t)row * N + col] = v;
            }
        }
    }
}

// ---------------------------------------------------------------------------
// Sequential complex selective scan.
// One wave32 per (batch b, channel e); lane = state index s (D_STATE==32).
// Per step: 1 exp + 1 sin + 1 cos per lane, coalesced B/C loads, scalar
// dt/x/z preloaded 32 timesteps at a time and broadcast via lane shuffles.
// Fuses  y = scan + D*x, then y *= silu(z), writes f16 for out-proj WMMA.
// ---------------------------------------------------------------------------
__global__ __launch_bounds__(256)
void scan_kernel(const float* __restrict__ dt,      // [MROWS][EE]
                 const float* __restrict__ xz,      // [MROWS][2*EE] (x_ssm|z)
                 const float* __restrict__ xdbl,    // [MROWS][128] (dt|B|C)
                 const float* __restrict__ A_log_real,  // [EE][NS]
                 const float* __restrict__ A_imag,      // [EE][NS]
                 const float* __restrict__ D_param,     // [EE]
                 _Float16* __restrict__ y_h) {          // [MROWS][EE]
    int gw   = (int)((blockIdx.x * (size_t)blockDim.x + threadIdx.x) >> 5);
    int lane = threadIdx.x & 31;
    int b = gw / EE;
    int e = gw - b * EE;

    float Ar = -__expf(A_log_real[(size_t)e * NS + lane]);
    float Ai = A_imag[(size_t)e * NS + lane];
    float Dp = D_param[e];
    float hr = 0.f, hi = 0.f;

    for (int t0 = 0; t0 < TT; t0 += 32) {
        int mpre = b * TT + t0 + lane;
        float dt_pre = dt[(size_t)mpre * EE + e];
        float x_pre  = xz[(size_t)mpre * (2 * EE) + e];
        float z_pre  = xz[(size_t)mpre * (2 * EE) + EE + e];
#pragma unroll 4
        for (int u = 0; u < 32; ++u) {
            float dtv = __shfl(dt_pre, u, 32);
            float xv  = __shfl(x_pre,  u, 32);
            float zv  = __shfl(z_pre,  u, 32);
            int m2 = b * TT + t0 + u;
            float Bv = xdbl[(size_t)m2 * 128 + 64 + lane];
            float Cv = xdbl[(size_t)m2 * 128 + 96 + lane];

            float sc  = __expf(Ar * dtv);
            float ang = Ai * dtv;
            float dAr = sc * __cosf(ang);
            float dAi = sc * __sinf(ang);
            float dBx = dtv * xv * Bv;
            float hr2 = dAr * hr - dAi * hi + dBx;
            hi = dAr * hi + dAi * hr;
            hr = hr2;

            float p = hr * Cv;
            for (int off = 16; off >= 1; off >>= 1)
                p += __shfl_xor(p, off, 32);
            if (lane == 0) {
                float y  = p + Dp * xv;
                float sz = zv / (1.f + __expf(-zv));     // silu gate
                y_h[(size_t)m2 * EE + e] = (_Float16)(y * sz);
            }
        }
    }
}

// ---------------------------------------------------------------------------
// Host launch
// ---------------------------------------------------------------------------
extern "C" void kernel_launch(void* const* d_in, const int* in_sizes, int n_in,
                              void* d_out, int out_size, void* d_ws, size_t ws_size,
                              hipStream_t stream) {
    const float* x          = (const float*)d_in[0];   // [2,2048,1024]
    const float* W_in       = (const float*)d_in[1];   // [4096,1024]
    const float* W_x        = (const float*)d_in[2];   // [128,2048]
    const float* W_dt       = (const float*)d_in[3];   // [2048,64]
    const float* b_dt       = (const float*)d_in[4];   // [2048]
    const float* A_log_real = (const float*)d_in[5];   // [2048,32]
    const float* A_imag     = (const float*)d_in[6];   // [2048,32]
    const float* D_param    = (const float*)d_in[7];   // [2048]
    const float* W_out      = (const float*)d_in[8];   // [1024,2048]
    const float* ln_g       = (const float*)d_in[9];
    const float* ln_b       = (const float*)d_in[10];
    float* out = (float*)d_out;                        // [2,2048,1024]

    // workspace carve-up (256B-aligned offsets)
    char* w = (char*)d_ws;
    auto carve = [&](size_t bytes) {
        char* p = w;
        w += (bytes + 255) & ~(size_t)255;
        return p;
    };
    _Float16* W_in_h  = (_Float16*)carve((size_t)2 * EE * DM * 2);   //  8 MB
    _Float16* W_x_h   = (_Float16*)carve((size_t)128 * EE * 2);      // .5 MB
    _Float16* W_dt_h  = (_Float16*)carve((size_t)EE * DTR * 2);      // .25MB
    _Float16* W_out_h = (_Float16*)carve((size_t)DM * EE * 2);       //  4 MB
    _Float16* xn_h    = (_Float16*)carve((size_t)MROWS * DM * 2);    //  8 MB
    float*    xz      = (float*)   carve((size_t)MROWS * 2*EE * 4);  // 64 MB
    _Float16* xssm_h  = (_Float16*)carve((size_t)MROWS * EE * 2);    // 16 MB
    float*    xdbl    = (float*)   carve((size_t)MROWS * 128 * 4);   //  2 MB
    _Float16* dtlow_h = (_Float16*)carve((size_t)MROWS * DTR * 2);   // .5 MB
    float*    dtbuf   = (float*)   carve((size_t)MROWS * EE * 4);    // 32 MB
    _Float16* y_h     = (_Float16*)carve((size_t)MROWS * EE * 2);    // 16 MB

    // 1) weights -> f16
    {
        size_t n;
        n = (size_t)2 * EE * DM;
        cvt_f32_f16_kernel<<<(n + 255) / 256, 256, 0, stream>>>(W_in, W_in_h, n);
        n = (size_t)128 * EE;
        cvt_f32_f16_kernel<<<(n + 255) / 256, 256, 0, stream>>>(W_x, W_x_h, n);
        n = (size_t)EE * DTR;
        cvt_f32_f16_kernel<<<(n + 255) / 256, 256, 0, stream>>>(W_dt, W_dt_h, n);
        n = (size_t)DM * EE;
        cvt_f32_f16_kernel<<<(n + 255) / 256, 256, 0, stream>>>(W_out, W_out_h, n);
    }

    // 2) LayerNorm -> xn_h (f16)
    layernorm_kernel<<<MROWS, 256, 0, stream>>>(x, ln_g, ln_b, xn_h);

    // 3) in-projection: xz = xn @ W_in^T (M=4096, N=4096, K=1024)
    //    fused epilogue also writes x_ssm (cols < 2048) as f16
    {
        int M = MROWS, N = 2 * EE, NG = N / 64;
        int blocks = (M / 16) * NG / 8;
        gemm_wmma_kernel<4, 4><<<blocks, 256, 0, stream>>>(
            xn_h, W_in_h, xz, M, N, 1024, nullptr, xssm_h);
    }

    // 4) x_dbl = x_ssm @ W_x^T  (M=4096, N=128, K=2048); also dt_low -> f16
    {
        int M = MROWS, N = 128, NG = N / 64;
        int blocks = (M / 16) * NG / 8;
        gemm_wmma_kernel<4, 2><<<blocks, 256, 0, stream>>>(
            xssm_h, W_x_h, xdbl, M, N, EE, nullptr, dtlow_h);
    }

    // 5) dt = softplus(dt_low @ W_dt^T + b_dt)  (M=4096, N=2048, K=64)
    {
        int M = MROWS, N = EE, NG = N / 64;
        int blocks = (M / 16) * NG / 8;
        gemm_wmma_kernel<4, 1><<<blocks, 256, 0, stream>>>(
            dtlow_h, W_dt_h, dtbuf, M, N, DTR, b_dt, nullptr);
    }

    // 6) sequential complex scan + D*x + silu(z) gate -> y_h (f16)
    {
        int waves  = BB * EE;              // 4096 waves, one per (b, channel)
        int blocks = waves * 32 / 256;     // 512
        scan_kernel<<<blocks, 256, 0, stream>>>(
            dtbuf, xz, xdbl, A_log_real, A_imag, D_param, y_h);
    }

    // 7) out = residual + y @ W_out^T  (M=4096, N=1024, K=2048)
    {
        int M = MROWS, N = DM, NG = N / 64;
        int blocks = (M / 16) * NG / 8;
        gemm_wmma_kernel<4, 3><<<blocks, 256, 0, stream>>>(
            y_h, W_out_h, out, M, N, EE, x, nullptr);
    }
}
